// CustomLSTM_11862699671824
// MI455X (gfx1250) — matrix operand order repaired and checked by reading, not compiled, not run on hardware
//
#include <hip/hip_runtime.h>
#include <stdint.h>

// ---------------------------------------------------------------------------
// CDNA5 (gfx1250) LSTM forward.
//   x_proj = X@W + b   : bf16 WMMA GEMM, fp32 accumulate
//   recurrence         : single persistent kernel, 512 steps, grid barrier,
//                        h double-buffered in bf16, U L2-resident (8MB bf16)
// WMMA: v_wmma_f32_16x16x32_bf16, 16x16 tiles.
// ---------------------------------------------------------------------------

typedef __attribute__((ext_vector_type(16))) __bf16 v16bf;
typedef __attribute__((ext_vector_type(8)))  float  v8f;

#define UNITS   1024
#define IDIM    512
#define BATCH   32
#define TSTEPS  512
#define NCOLS   4096          // 4*UNITS
#define NTILES  256           // 4096/16 column tiles
#define KT_W    16            // 512/32  k-tiles for W
#define KT_U    32            // 1024/32 k-tiles for U
#define RBLOCKS 64            // persistent recurrence blocks (one per 16 hidden cols)

static __device__ __forceinline__ unsigned short f32_to_bf16(float f) {
  unsigned int u = __builtin_bit_cast(unsigned int, f);
  u += 0x7FFFu + ((u >> 16) & 1u);            // round-to-nearest-even
  return (unsigned short)(u >> 16);
}

union AFrag { uint4  q[2]; v16bf v; };        // 16x32 bf16 A / 32x16 bf16 B fragment
union CFrag { float4 q[2]; v8f   v; float f[8]; };

// ---------------------------------------------------------------------------
// Init: zero h (both buffers), c, barrier words.
// ---------------------------------------------------------------------------
__global__ void init_kernel(unsigned short* hbf, float* cst, unsigned* bar) {
  int i = blockIdx.x * blockDim.x + threadIdx.x;   // 65536 threads
  if (i < 2 * BATCH * UNITS) hbf[i] = 0;
  if (i < BATCH * UNITS)     cst[i] = 0.f;
  if (i < 2)                 bar[i] = 0u;
}

// ---------------------------------------------------------------------------
// Pack a row-major fp32 [K, 4096] matrix into per-lane B-fragment order (bf16).
// Tile (k_tile, n_tile) holds 512 bf16:  elem = lane*16 + j,
//   lanes 0-15 : N = n0+lane,    K = k0 + j        (j = 0..15)
//   lanes16-31 : N = n0+lane-16, K = k0 + 16 + j
// so each lane's 16 operand values are 32 contiguous bytes.
// ---------------------------------------------------------------------------
__global__ void pack_b_kernel(const float* __restrict__ src,
                              unsigned short* __restrict__ dst,
                              int k_tiles) {
  int idx = blockIdx.x * blockDim.x + threadIdx.x;
  int total = k_tiles * NTILES * 512;
  if (idx >= total) return;
  int tile   = idx >> 9;
  int within = idx & 511;
  int lane   = within >> 4;
  int j      = within & 15;
  int n_tile = tile & (NTILES - 1);
  int k_tile = tile >> 8;
  int k = k_tile * 32 + ((lane >> 4) << 4) + j;
  int n = n_tile * 16 + (lane & 15);
  dst[idx] = f32_to_bf16(src[(size_t)k * NCOLS + n]);
}

// ---------------------------------------------------------------------------
// fp32 -> bf16 elementwise (inputs X stay row-major [B,T,I]).
// ---------------------------------------------------------------------------
__global__ void cvt_bf16_kernel(const float* __restrict__ src,
                                unsigned short* __restrict__ dst, int n) {
  int i = blockIdx.x * blockDim.x + threadIdx.x;
  if (i < n) dst[i] = f32_to_bf16(src[i]);
}

// ---------------------------------------------------------------------------
// x_proj GEMM: for each t, Xbf_t[32,512] @ W[512,4096] + b.
// grid = (64, 512): blockIdx.y = t; 8 waves/block, wave -> one 16x16 tile.
// Output stored fragment-ordered: xp[t][m_tile(2)][n_tile(256)][lane(32)][reg(8)]
// ---------------------------------------------------------------------------
__global__ void __launch_bounds__(256)
xproj_kernel(const unsigned short* __restrict__ Xbf,
             const unsigned short* __restrict__ Wp,
             const float* __restrict__ bias,
             float* __restrict__ xp) {
  const int t      = blockIdx.y;
  const int lane   = threadIdx.x & 31;
  const int w      = threadIdx.x >> 5;
  const int id     = blockIdx.x * 8 + w;     // 0..511
  const int m_tile = id & 1;
  const int n_tile = id >> 1;                // 0..255
  const int l16    = lane & 15;
  const int hi     = lane >> 4;

  // A: rows are batch entries m0+l16 at time t; lane-contiguous 32B k-run (k = 16*hi + s).
  const unsigned short* arow =
      Xbf + ((size_t)(m_tile * 16 + l16) * TSTEPS + t) * IDIM + hi * 16;
  const unsigned short* bptr = Wp + (size_t)n_tile * 512 + lane * 16;

  CFrag acc; acc.v = (v8f){};
#pragma unroll
  for (int kt = 0; kt < KT_W; ++kt) {
    AFrag a, b;
    a.q[0] = *(const uint4*)(arow + kt * 32);
    a.q[1] = *(const uint4*)(arow + kt * 32 + 8);
    const uint4* bp = (const uint4*)(bptr + (size_t)kt * (NTILES * 512));
    b.q[0] = bp[0];
    b.q[1] = bp[1];
    acc.v = __builtin_amdgcn_wmma_f32_16x16x32_bf16(
        false, a.v, false, b.v, (short)0, acc.v, false, false);
  }
  const float bv = bias[n_tile * 16 + l16];   // N depends only on lane
#pragma unroll
  for (int r = 0; r < 8; ++r) acc.f[r] += bv;

  float* outp = xp + ((((size_t)t * 2 + m_tile) * NTILES + n_tile) * 32 + lane) * 8;
  ((float4*)outp)[0] = acc.q[0];
  ((float4*)outp)[1] = acc.q[1];
}

// ---------------------------------------------------------------------------
// Persistent recurrence: 64 blocks (j = hidden col tile), 8 waves each:
//   wave w: m_tile = w&1 (batch rows 0-15 / 16-31), gate = w>>1 (i,f,g,o)
// Per step: acc = xp tile; 32x WMMA over K=1024 of h@U; LDS exchange of the
// 8 z tiles; gates + c/h update; device-wide barrier (h double-buffered).
// ---------------------------------------------------------------------------
__global__ void __launch_bounds__(256)
lstm_recur_kernel(const float* __restrict__ xp,
                  const unsigned short* __restrict__ Up,
                  unsigned short* __restrict__ hbf,   // 2 x [32,1024] bf16
                  float* __restrict__ cst,            // [32,1024]
                  float* __restrict__ out,            // [32,512,1024]
                  unsigned* __restrict__ bar) {
  __shared__ float zs[8][16][16];                     // [wave][N][M]

  const int lane   = threadIdx.x & 31;
  const int w      = threadIdx.x >> 5;
  const int m_tile = w & 1;
  const int gate   = w >> 1;
  const int j      = blockIdx.x;                      // 0..63
  const int n_tile = gate * 64 + j;
  const int l16    = lane & 15;
  const int hi     = lane >> 4;

  const unsigned short* bbase = Up + (size_t)n_tile * 512 + lane * 16;

  // gate-phase per-thread mapping: element (batch eb, col ecol & ecol+8)
  const int eb   = threadIdx.x & 31;
  const int ecol = threadIdx.x >> 5;
  const int emt  = eb >> 4;
  const int em   = eb & 15;

  for (int t = 0; t < TSTEPS; ++t) {
    const unsigned short* hread  = hbf + (size_t)(t & 1) * (BATCH * UNITS);
    unsigned short*       hwrite = (unsigned short*)hbf +
                                   (size_t)((t & 1) ^ 1) * (BATCH * UNITS);

    // accumulator = x_proj tile (fragment-ordered, 32B/lane)
    CFrag acc;
    const float* xptile =
        xp + ((((size_t)t * 2 + m_tile) * NTILES + n_tile) * 32 + lane) * 8;
    acc.q[0] = ((const float4*)xptile)[0];
    acc.q[1] = ((const float4*)xptile)[1];

    const unsigned short* arow =
        hread + (size_t)(m_tile * 16 + l16) * UNITS + hi * 16;
#pragma unroll 4
    for (int kt = 0; kt < KT_U; ++kt) {
      AFrag a, b;
      a.q[0] = *(const uint4*)(arow + kt * 32);
      a.q[1] = *(const uint4*)(arow + kt * 32 + 8);
      const uint4* bp = (const uint4*)(bbase + (size_t)kt * (NTILES * 512));
      b.q[0] = bp[0];
      b.q[1] = bp[1];
      acc.v = __builtin_amdgcn_wmma_f32_16x16x32_bf16(
          false, a.v, false, b.v, (short)0, acc.v, false, false);
    }

    // z tile -> LDS transposed [w][N][M]: lane owns column N=l16, rows hi*8..+7
    {
      float4* d0 = (float4*)&zs[w][l16][hi * 8];
      d0[0] = acc.q[0];
      d0[1] = acc.q[1];
    }
    __syncthreads();

    // gates: 512 (b,col) elements, 2 per thread
#pragma unroll
    for (int p = 0; p < 2; ++p) {
      const int col = ecol + p * 8;
      const float zi = zs[0 + emt][col][em];
      const float zf = zs[2 + emt][col][em];
      const float zg = zs[4 + emt][col][em];
      const float zo = zs[6 + emt][col][em];
      const float ig = 1.f / (1.f + __expf(-zi));
      const float fg = 1.f / (1.f + __expf(-zf));
      const float gg = tanhf(zg);
      const float og = 1.f / (1.f + __expf(-zo));
      const int cidx = eb * UNITS + j * 16 + col;     // owned exclusively by block j
      const float cn = fg * cst[cidx] + ig * gg;
      const float hn = og * tanhf(cn);
      cst[cidx]    = cn;
      hwrite[cidx] = f32_to_bf16(hn);
      out[((size_t)eb * TSTEPS + t) * UNITS + j * 16 + col] = hn;
    }
    __syncthreads();

    // device-wide barrier (sense-reversing, generation counter)
    __threadfence();                                   // flush h writes (release)
    if (threadIdx.x == 0) {
      unsigned g = atomicAdd(&bar[1], 0u);             // read generation
      if (atomicAdd(&bar[0], 1u) == RBLOCKS - 1) {
        atomicExch(&bar[0], 0u);
        atomicAdd(&bar[1], 1u);
      } else {
        while (atomicAdd(&bar[1], 0u) == g) __builtin_amdgcn_s_sleep(2);
      }
    }
    __syncthreads();
    __threadfence();                                   // acquire: invalidate stale h
  }
}

// ---------------------------------------------------------------------------
extern "C" void kernel_launch(void* const* d_in, const int* in_sizes, int n_in,
                              void* d_out, int out_size, void* d_ws, size_t ws_size,
                              hipStream_t stream) {
  const float* X    = (const float*)d_in[0];   // [32,512,512]
  const float* W    = (const float*)d_in[1];   // [512,4096]
  const float* U    = (const float*)d_in[2];   // [1024,4096]
  const float* bias = (const float*)d_in[3];   // [4096]
  float* out = (float*)d_out;

  char* ws = (char*)d_ws;
  size_t off = 0;
  auto alloc = [&](size_t bytes) {
    size_t p = off;
    off += (bytes + 255) & ~(size_t)255;
    return p;
  };
  unsigned short* Wp  = (unsigned short*)(ws + alloc((size_t)IDIM  * NCOLS * 2));
  unsigned short* Up  = (unsigned short*)(ws + alloc((size_t)UNITS * NCOLS * 2));
  unsigned short* Xbf = (unsigned short*)(ws + alloc((size_t)BATCH * TSTEPS * IDIM * 2));
  float*          xp  = (float*)(ws + alloc((size_t)BATCH * TSTEPS * NCOLS * 4));
  unsigned short* hbf = (unsigned short*)(ws + alloc((size_t)2 * BATCH * UNITS * 2));
  float*          cst = (float*)(ws + alloc((size_t)BATCH * UNITS * 4));
  unsigned*       bar = (unsigned*)(ws + alloc(256));
  (void)ws_size; (void)in_sizes; (void)n_in; (void)out_size;

  init_kernel<<<dim3(256), 256, 0, stream>>>(hbf, cst, bar);
  pack_b_kernel<<<(IDIM  * NCOLS + 255) / 256, 256, 0, stream>>>(W, Wp, KT_W);
  pack_b_kernel<<<(UNITS * NCOLS + 255) / 256, 256, 0, stream>>>(U, Up, KT_U);
  cvt_bf16_kernel<<<(BATCH * TSTEPS * IDIM + 255) / 256, 256, 0, stream>>>(
      X, Xbf, BATCH * TSTEPS * IDIM);
  xproj_kernel<<<dim3(64, TSTEPS), 256, 0, stream>>>(Xbf, Wp, bias, xp);
  lstm_recur_kernel<<<RBLOCKS, 256, 0, stream>>>(xp, Up, hbf, cst, out, bar);
}
